// GRU_MIL_Siamese_29738353558314
// MI455X (gfx1250) — matrix operand order repaired and checked by reading, not compiled, not run on hardware
//
#include <hip/hip_runtime.h>
#include <hip/hip_bf16.h>

typedef _Float16 v16h __attribute__((ext_vector_type(16)));
typedef _Float16 h8   __attribute__((ext_vector_type(8)));
typedef float    v8f  __attribute__((ext_vector_type(8)));

union V16H { v16h v; _Float16 e[16]; };
union V8F  { v8f v; float e[8]; };

#define TLEN 1000
#define HID  64
#define NB   128   // batch
#define NW   11    // windows

__device__ __forceinline__ float fexp(float x) {
  return __builtin_amdgcn_exp2f(x * 1.4426950408889634f);
}
__device__ __forceinline__ float sigm(float x) {
  return __builtin_amdgcn_rcpf(1.0f + fexp(-x));
}
__device__ __forceinline__ float tanh_f(float x) {
  return 1.0f - 2.0f * __builtin_amdgcn_rcpf(fexp(2.0f * x) + 1.0f);
}

__device__ __forceinline__ v8f wm(v16h a, v16h b, v8f c) {
  // D = A(16x32 f16) * B(32x16 f16) + C(16x16 f32)
  return __builtin_amdgcn_wmma_f32_16x16x32_f16(
      /*neg_a=*/false, a, /*neg_b=*/false, b,
      /*c_mod=*/(short)0, c, /*reuse_a=*/false, /*reuse_b=*/false);
}

// -------- Phase 1: batched GRU over 1408 sequences, WMMA recurrence --------
// One workgroup = 16 sequences, 4 waves; wave w owns hidden/gate rows 16w..16w+15.
__global__ __launch_bounds__(128)
void gru_phase1(const float* __restrict__ x,
                const float* __restrict__ w_ih,
                const float* __restrict__ w_hh,
                const float* __restrict__ b_ih,
                const float* __restrict__ b_hh,
                float* __restrict__ emb) {
  __shared__ __align__(16) _Float16 hbuf[2][16 * HID];  // [seq][hid], double buffered

  const int tid  = threadIdx.x;
  const int wave = tid >> 5;
  const int lane = tid & 31;
  const int lo   = lane & 15;   // D-layout: gate column / A-layout: seq row
  const int hi   = lane >> 4;   // half select
  const int wg   = blockIdx.x;

  for (int i = tid; i < 16 * HID; i += 128) hbuf[0][i] = (_Float16)0.0f;

  const int g0r = 16 * wave;          // r-gate rows in w_hh
  const int g0z = 64 + 16 * wave;     // z-gate rows
  const int g0n = 128 + 16 * wave;    // n-gate rows

  // ---- B operands (K x 16 gate columns) in WMMA B-layout, f16, resident in VGPRs ----
  V16H Br0, Br1, Bz0, Bz1, Bn0, Bn1, B2r, B2z, B2nh, B2nx, A2;
#pragma unroll
  for (int p = 0; p < 8; ++p) {
    const int kk = 2 * p + 16 * hi;  // K row within 32-chunk held by this lane/VGPR
    Br0.e[2*p]   = (_Float16)w_hh[(g0r + lo) * HID + kk];
    Br0.e[2*p+1] = (_Float16)w_hh[(g0r + lo) * HID + kk + 1];
    Br1.e[2*p]   = (_Float16)w_hh[(g0r + lo) * HID + 32 + kk];
    Br1.e[2*p+1] = (_Float16)w_hh[(g0r + lo) * HID + 32 + kk + 1];
    Bz0.e[2*p]   = (_Float16)w_hh[(g0z + lo) * HID + kk];
    Bz0.e[2*p+1] = (_Float16)w_hh[(g0z + lo) * HID + kk + 1];
    Bz1.e[2*p]   = (_Float16)w_hh[(g0z + lo) * HID + 32 + kk];
    Bz1.e[2*p+1] = (_Float16)w_hh[(g0z + lo) * HID + 32 + kk + 1];
    Bn0.e[2*p]   = (_Float16)w_hh[(g0n + lo) * HID + kk];
    Bn0.e[2*p+1] = (_Float16)w_hh[(g0n + lo) * HID + kk + 1];
    Bn1.e[2*p]   = (_Float16)w_hh[(g0n + lo) * HID + 32 + kk];
    Bn1.e[2*p+1] = (_Float16)w_hh[(g0n + lo) * HID + 32 + kk + 1];
  }
  // Extra K-chunk: A2 = [x_t @K0, 1.0 @K1, 0...]; B2 rows K0=w_ih, K1=bias.
#pragma unroll
  for (int i = 0; i < 16; ++i) {
    B2r.e[i] = (_Float16)0.0f; B2z.e[i] = (_Float16)0.0f;
    B2nh.e[i] = (_Float16)0.0f; B2nx.e[i] = (_Float16)0.0f;
    A2.e[i] = (_Float16)0.0f;
  }
  if (hi == 0) {  // only K rows 0,1 are nonzero -> lanes 0..15, VGPR 0
    B2r.e[0]  = (_Float16)w_ih[g0r + lo];
    B2r.e[1]  = (_Float16)(b_ih[g0r + lo] + b_hh[g0r + lo]);
    B2z.e[0]  = (_Float16)w_ih[g0z + lo];
    B2z.e[1]  = (_Float16)(b_ih[g0z + lo] + b_hh[g0z + lo]);
    B2nh.e[1] = (_Float16)b_hh[g0n + lo];            // hn bias (inside r*hn)
    B2nx.e[0] = (_Float16)w_ih[g0n + lo];            // xn = x*w_ih_n + b_ih_n
    B2nx.e[1] = (_Float16)b_ih[g0n + lo];
    A2.e[1]   = (_Float16)1.0f;                      // constant-1 K lane for biases
  }

  float hstate[8];
#pragma unroll
  for (int v = 0; v < 8; ++v) hstate[v] = 0.0f;

  const float* xseq = x + (size_t)(wg * 16 + lo) * TLEN;

  __syncthreads();

  for (int tb = 0; tb < TLEN; tb += 8) {
    const float4 xa = *(const float4*)(xseq + tb);
    const float4 xb = *(const float4*)(xseq + tb + 4);
    if (tb + 8 < TLEN) __builtin_prefetch(xseq + tb + 8, 0, 3);  // global_prefetch_b8
    const float xq[8] = {xa.x, xa.y, xa.z, xa.w, xb.x, xb.y, xb.z, xb.w};

#pragma unroll
    for (int tt = 0; tt < 8; ++tt) {
      const int cur = tt & 1;       // tb multiple of 8 keeps parity
      const int nxt = cur ^ 1;

      // A operands = h in WMMA A-layout, from LDS (ds_load_b128 x4)
      V16H A0, A1;
      const _Float16* hrow = &hbuf[cur][lo * HID];
      *(h8*)&A0.e[0] = *(const h8*)(hrow + 16 * hi);
      *(h8*)&A0.e[8] = *(const h8*)(hrow + 8 + 16 * hi);
      *(h8*)&A1.e[0] = *(const h8*)(hrow + 32 + 16 * hi);
      *(h8*)&A1.e[8] = *(const h8*)(hrow + 40 + 16 * hi);
      A2.e[0] = (hi == 0) ? (_Float16)xq[tt] : (_Float16)0.0f;

      const v8f zero = {};
      v8f accr = wm(A2.v, B2r.v, zero);
      accr     = wm(A0.v, Br0.v, accr);
      accr     = wm(A1.v, Br1.v, accr);
      v8f accz = wm(A2.v, B2z.v, zero);
      accz     = wm(A0.v, Bz0.v, accz);
      accz     = wm(A1.v, Bz1.v, accz);
      v8f acch = wm(A2.v, B2nh.v, zero);   // hn = h@w_hh_n^T + b_hh_n
      acch     = wm(A0.v, Bn0.v, acch);
      acch     = wm(A1.v, Bn1.v, acch);
      v8f accx = wm(A2.v, B2nx.v, zero);   // xn = x*w_ih_n + b_ih_n

      V8F Ur, Uz, Uh, Ux;
      Ur.v = accr; Uz.v = accz; Uh.v = acch; Ux.v = accx;
#pragma unroll
      for (int v = 0; v < 8; ++v) {
        const float r = sigm(Ur.e[v]);
        const float z = sigm(Uz.e[v]);
        const float n = tanh_f(Ux.e[v] + r * Uh.e[v]);
        const float h = (1.0f - z) * n + z * hstate[v];
        hstate[v] = h;
        // D-layout: lane holds seq row v+8*hi, hidden col 16*wave+lo
        hbuf[nxt][(v + 8 * hi) * HID + 16 * wave + lo] = (_Float16)h;
      }
      __syncthreads();
    }
  }

#pragma unroll
  for (int v = 0; v < 8; ++v)
    emb[(size_t)(wg * 16 + v + 8 * hi) * HID + 16 * wave + lo] = hstate[v];
}

// -------- Phase 2: attention MLP + softmax + aggregate + FC (tiny) --------
__global__ __launch_bounds__(64)
void attn_phase2(const float* __restrict__ emb,
                 const float* __restrict__ aw1, const float* __restrict__ ab1,
                 const float* __restrict__ aw2, const float* __restrict__ ab2,
                 const float* __restrict__ fcw, const float* __restrict__ fcb,
                 float* __restrict__ out) {
  __shared__ float E[NW][HID];
  __shared__ float T1[NW][HID];
  __shared__ float SC[NW];
  __shared__ float AG[HID];

  const int b = blockIdx.x;
  const int j = threadIdx.x;  // 0..63

#pragma unroll
  for (int i = 0; i < NW; ++i) E[i][j] = emb[(size_t)(b * NW + i) * HID + j];
  __syncthreads();

  for (int i = 0; i < NW; ++i) {
    float s = ab1[j];
#pragma unroll
    for (int k = 0; k < HID; ++k) s += E[i][k] * aw1[j * HID + k];
    T1[i][j] = tanh_f(s);
  }
  __syncthreads();

  if (j < NW) {
    float s = ab2[0];
#pragma unroll
    for (int k = 0; k < HID; ++k) s += T1[j][k] * aw2[k];
    SC[j] = s;
  }
  __syncthreads();

  // softmax over windows (redundant per thread; 11 elements)
  float m = SC[0];
#pragma unroll
  for (int i = 1; i < NW; ++i) m = fmaxf(m, SC[i]);
  float w[NW], sum = 0.0f;
#pragma unroll
  for (int i = 0; i < NW; ++i) { w[i] = fexp(SC[i] - m); sum += w[i]; }
  const float inv = 1.0f / sum;

  float agg = 0.0f;
#pragma unroll
  for (int i = 0; i < NW; ++i) agg += (w[i] * inv) * E[i][j];
  AG[j] = agg;
  if (j < NW) out[NB * HID + b * NW + j] = w[j] * inv;  // weights output
  __syncthreads();

  float s = fcb[j];
#pragma unroll
  for (int k = 0; k < HID; ++k) s += AG[k] * fcw[j * HID + k];
  out[b * HID + j] = s;  // final_emb output
}

extern "C" void kernel_launch(void* const* d_in, const int* in_sizes, int n_in,
                              void* d_out, int out_size, void* d_ws, size_t ws_size,
                              hipStream_t stream) {
  const float* x    = (const float*)d_in[0];
  const float* w_ih = (const float*)d_in[1];
  const float* w_hh = (const float*)d_in[2];
  const float* b_ih = (const float*)d_in[3];
  const float* b_hh = (const float*)d_in[4];
  const float* aw1  = (const float*)d_in[5];
  const float* ab1  = (const float*)d_in[6];
  const float* aw2  = (const float*)d_in[7];
  const float* ab2  = (const float*)d_in[8];
  const float* fcw  = (const float*)d_in[9];
  const float* fcb  = (const float*)d_in[10];
  float* out = (float*)d_out;
  float* emb = (float*)d_ws;  // 1408 * 64 floats of scratch

  gru_phase1<<<88, 128, 0, stream>>>(x, w_ih, w_hh, b_ih, b_hh, emb);
  attn_phase2<<<NB, HID, 0, stream>>>(emb, aw1, ab1, aw2, ab2, fcw, fcb, out);
}
